// MutualInfoMetric_61314953118511
// MI455X (gfx1250) — compile-verified, matrix operand order checked by reading, not run on hardware
//
#include <hip/hip_runtime.h>
#include <hip/hip_bf16.h>
#include <math.h>

typedef __attribute__((ext_vector_type(16))) _Float16     v16h;
typedef __attribute__((ext_vector_type(8)))  float        v8f;
typedef __attribute__((ext_vector_type(4)))  unsigned int v4u;
typedef __attribute__((ext_vector_type(8)))  int          v8i;
typedef __attribute__((ext_vector_type(4)))  int          v4i;

// ---------------------------------------------------------------------------
// Kernel 1: conv1 (1->64, 3x3 SAME) + bias + relu + 2x2 maxpool
//   inputs : 4 set pointers, each [4,1,1,40,40] f32
//   output : P1 [16 img, 64 ch, 20, 20] f32
// ---------------------------------------------------------------------------
__global__ void enc_conv1_pool(const float* __restrict__ s0, const float* __restrict__ s1,
                               const float* __restrict__ s2, const float* __restrict__ s3,
                               const float* __restrict__ cw1, const float* __restrict__ cb1,
                               float* __restrict__ P1) {
    int idx = blockIdx.x * blockDim.x + threadIdx.x;
    if (idx >= 16 * 64 * 20 * 20) return;
    int pw  = idx % 20;
    int ph  = (idx / 20) % 20;
    int oc  = (idx / 400) % 64;
    int img = idx / (400 * 64);

    const float* x;
    {
        int set = img >> 2, sub = img & 3;
        const float* base = (set == 0) ? s0 : (set == 1) ? s1 : (set == 2) ? s2 : s3;
        x = base + sub * 1600;
    }

    float w[9];
#pragma unroll
    for (int t = 0; t < 9; ++t) w[t] = cw1[oc * 9 + t];
    float bias = cb1[oc];

    float best = -INFINITY;
#pragma unroll
    for (int py = 0; py < 2; ++py)
#pragma unroll
    for (int px = 0; px < 2; ++px) {
        int y = 2 * ph + py, xx = 2 * pw + px;
        float acc = bias;
#pragma unroll
        for (int dy = 0; dy < 3; ++dy)
#pragma unroll
        for (int dx = 0; dx < 3; ++dx) {
            int iy = y + dy - 1, ix = xx + dx - 1;
            float v = (iy >= 0 && iy < 40 && ix >= 0 && ix < 40) ? x[iy * 40 + ix] : 0.0f;
            acc = fmaf(v, w[dy * 3 + dx], acc);
        }
        best = fmaxf(best, fmaxf(acc, 0.0f));
    }
    P1[idx] = best;   // idx == ((img*64 + oc)*400 + ph*20 + pw)
}

// ---------------------------------------------------------------------------
// Kernel 2: conv2 (64->64, 3x3 SAME) + bias + relu + 2x2 maxpool, flatten
//   input  : P1 [16,64,20,20]     output : E [16, 6400] (c*100 + h*10 + w)
//   block = one (img, oc); weights for oc staged through LDS
// ---------------------------------------------------------------------------
__global__ void enc_conv2_pool(const float* __restrict__ P1, const float* __restrict__ cw2,
                               const float* __restrict__ cb2, float* __restrict__ E) {
    __shared__ float wts[576];
    int img = blockIdx.x >> 6;
    int oc  = blockIdx.x & 63;
    for (int t = threadIdx.x; t < 576; t += blockDim.x)
        wts[t] = cw2[oc * 576 + t];
    __syncthreads();

    int t = threadIdx.x;
    if (t >= 100) return;
    int ph = t / 10, pw = t % 10;
    const float* xin = P1 + (size_t)img * 64 * 400;

    float c00 = 0.f, c01 = 0.f, c10 = 0.f, c11 = 0.f;
    for (int ic = 0; ic < 64; ++ic) {
        const float* ch = xin + ic * 400;
        float p[4][4];
#pragma unroll
        for (int r = 0; r < 4; ++r)
#pragma unroll
        for (int cidx = 0; cidx < 4; ++cidx) {
            int iy = 2 * ph - 1 + r, ix = 2 * pw - 1 + cidx;
            p[r][cidx] = (iy >= 0 && iy < 20 && ix >= 0 && ix < 20) ? ch[iy * 20 + ix] : 0.0f;
        }
        const float* w = &wts[ic * 9];
#pragma unroll
        for (int dy = 0; dy < 3; ++dy)
#pragma unroll
        for (int dx = 0; dx < 3; ++dx) {
            float wv = w[dy * 3 + dx];
            c00 = fmaf(p[dy][dx],         wv, c00);
            c01 = fmaf(p[dy][dx + 1],     wv, c01);
            c10 = fmaf(p[dy + 1][dx],     wv, c10);
            c11 = fmaf(p[dy + 1][dx + 1], wv, c11);
        }
    }
    float bias = cb2[oc];
    float v = fmaxf(fmaxf(fmaxf(c00 + bias, 0.f), fmaxf(c01 + bias, 0.f)),
                    fmaxf(fmaxf(c10 + bias, 0.f), fmaxf(c11 + bias, 0.f)));
    E[(size_t)img * 6400 + oc * 100 + ph * 10 + pw] = v;
}

// ---------------------------------------------------------------------------
// Kernel 3: H[16,1024] = E[16,6400] @ [W1a | W1b]
//   TDM (tensor_load_to_lds) double-buffers 256x16 f32 tiles of w1 into LDS,
//   8 waves each consume one 32-row k-step per chunk with
//   v_wmma_f32_16x16x32_f16, partials reduced across waves with ds_add_f32.
// ---------------------------------------------------------------------------
#define KCHUNK 256                  // w1 rows per TDM chunk
#define NCHUNK 25                   // 6400 / 256

__global__ void gemm_wmma(const float* __restrict__ E, const float* __restrict__ w1,
                          float* __restrict__ H) {
    __shared__ float bbuf[2][KCHUNK * 16];   // 2 x 16 KB double buffer
    __shared__ float cacc[256];

    int tile  = blockIdx.x;                                   // 0..63
    int cbase = (tile < 32) ? tile * 16 : (tile - 32) * 16;   // w1 column base
    int koff  = (tile < 32) ? 0 : 6400;                       // w1 row offset

    int lane = threadIdx.x & 31;
    int wave = threadIdx.x >> 5;                              // 0..7
    int m    = lane & 15;                                     // A row / B col
    int hi   = lane >> 4;                                     // K-stripe half select

    const float* Arow = E + (size_t)m * 6400;

    // ---- TDM descriptor (all block-uniform -> SGPRs) ----
    unsigned long long gbase =
        (unsigned long long)(uintptr_t)(w1 + (size_t)koff * 512 + cbase);
    unsigned int ldsa[2];
    ldsa[0] = (unsigned int)(uintptr_t)&bbuf[0][0];           // LDS byte offset (addr[31:0])
    ldsa[1] = (unsigned int)(uintptr_t)&bbuf[1][0];

    v8i g1;
    g1[0] = 2 << 16;                 // workgroup_mask=0 | data_size=2 (4 bytes)
    g1[1] = (int)(512u << 16);       // tensor_dim0[15:0]=512 in [63:48]
    g1[2] = (int)(12800u << 16);     // tensor_dim0[31:16]=0 | tensor_dim1[15:0]=12800
    g1[3] = 16 << 16;                // tensor_dim1[31:16]=0 | tile_dim0=16
    g1[4] = KCHUNK;                  // tile_dim1=256 | tile_dim2=0
    g1[5] = 512;                     // tensor_dim0_stride[31:0]=512
    g1[6] = 0;                       // stride hi | tensor_dim1_stride lo (2D: unused)
    g1[7] = 0;
    v4i gz4 = {0, 0, 0, 0};          // groups 2/3 unused (2D tensor)
    v8i gz8 = {0, 0, 0, 0, 0, 0, 0, 0};

    auto issue = [&](int ch) {
        unsigned long long ga = gbase + (unsigned long long)ch * (KCHUNK * 512ull * 4ull);
        v4u g0;
        g0[0] = 1u;                                              // count=1 (valid), user mode
        g0[1] = ldsa[ch & 1];                                    // lds_addr
        g0[2] = (unsigned int)ga;                                // global_addr[31:0]
        g0[3] = ((unsigned int)(ga >> 32) & 0x01FFFFFFu)         // global_addr[56:32]
              | (2u << 30);                                      // type=2 ("image")
        __builtin_amdgcn_tensor_load_to_lds(g0, g1, gz4, gz4, gz8, 0);
    };

    v8f c = {};
    if (wave == 0) issue(0);

    for (int ch = 0; ch < NCHUNK; ++ch) {
        if (wave == 0) {
            if (ch + 1 < NCHUNK) {
                issue(ch + 1);                                   // prefetch next chunk
                __builtin_amdgcn_s_wait_tensorcnt(1);            // chunk ch landed
            } else {
                __builtin_amdgcn_s_wait_tensorcnt(0);
            }
        }
        __syncthreads();                                         // chunk ch visible to all

        const float* B = &bbuf[ch & 1][wave * 32 * 16];          // this wave's 32 rows
        int kb = ch * KCHUNK + wave * 32;                        // global k base for A
        v16h a, b;
#pragma unroll
        for (int e = 0; e < 16; ++e) {
            // ISA 16-bit A/B layout: element e -> K = [e<8?0:16] + hi*8 + (e&7)
            int k = ((e < 8) ? 0 : 16) + hi * 8 + (e & 7);
            a[e] = (_Float16)Arow[kb + k];
            b[e] = (_Float16)B[k * 16 + m];
        }
        c = __builtin_amdgcn_wmma_f32_16x16x32_f16(
                /*neg_a=*/false, a, /*neg_b=*/false, b,
                /*c_mod=*/(short)0, c, /*reuse_a=*/false, /*reuse_b=*/false);

        __syncthreads();                                         // done reading buf[ch&1]
    }

    cacc[threadIdx.x] = 0.0f;
    __syncthreads();
#pragma unroll
    for (int r = 0; r < 8; ++r) {
        int mm = r + hi * 8;                                     // C/D: VGPR r holds row r(+8)
        atomicAdd(&cacc[mm * 16 + m], c[r]);
    }
    __syncthreads();

    int idx = threadIdx.x;                                       // 0..255 -> one H element
    int om = idx >> 4, on = idx & 15;
    H[(size_t)om * 1024 + tile * 16 + on] = cacc[idx];
}

// ---------------------------------------------------------------------------
// Kernel 4: for each pair p and (i,j): sigmoid( relu(HA[ai]+HB[bj]+b1) . w2 + b2 )
// ---------------------------------------------------------------------------
__global__ void pair_head(const float* __restrict__ H, const float* __restrict__ b1,
                          const float* __restrict__ w2, const float* __restrict__ b2,
                          float* __restrict__ out) {
    int p = blockIdx.x;
    int pa = (p < 4) ? p : (p - 4);
    int pb = (p < 4) ? p : (((p - 4) + 1) & 3);

    int wave = threadIdx.x >> 5;   // 0..15
    int lane = threadIdx.x & 31;
    int i = wave >> 2, j = wave & 3;

    const float* ra = H + (size_t)(pa * 4 + i) * 1024;           // HA row
    const float* rb = H + (size_t)(pb * 4 + j) * 1024 + 512;     // HB row

    float acc = 0.0f;
    for (int cc = lane; cc < 512; cc += 32) {
        float h = ra[cc] + rb[cc] + b1[cc];
        h = fmaxf(h, 0.0f);
        acc = fmaf(h, w2[cc], acc);
    }
#pragma unroll
    for (int off = 16; off > 0; off >>= 1)
        acc += __shfl_xor(acc, off, 32);

    if (lane == 0) {
        float z = acc + b2[0];
        out[p * 16 + i * 4 + j] = 1.0f / (1.0f + expf(-z));
    }
}

// ---------------------------------------------------------------------------
extern "C" void kernel_launch(void* const* d_in, const int* in_sizes, int n_in,
                              void* d_out, int out_size, void* d_ws, size_t ws_size,
                              hipStream_t stream) {
    const float* s0  = (const float*)d_in[0];
    const float* s1  = (const float*)d_in[1];
    const float* s2  = (const float*)d_in[2];
    const float* s3  = (const float*)d_in[3];
    const float* cw1 = (const float*)d_in[4];
    const float* cb1 = (const float*)d_in[5];
    const float* cw2 = (const float*)d_in[6];
    const float* cb2 = (const float*)d_in[7];
    const float* w1  = (const float*)d_in[8];
    const float* b1  = (const float*)d_in[9];
    const float* w2  = (const float*)d_in[10];
    const float* b2  = (const float*)d_in[11];
    float* out = (float*)d_out;

    // workspace layout (floats): P1[16*64*400] | E[16*6400] | H[16*1024]
    float* P1 = (float*)d_ws;
    float* E  = P1 + 16 * 64 * 400;
    float* H  = E + 16 * 6400;

    enc_conv1_pool<<<1600, 256, 0, stream>>>(s0, s1, s2, s3, cw1, cb1, P1);
    enc_conv2_pool<<<1024, 128, 0, stream>>>(P1, cw2, cb2, E);
    gemm_wmma<<<64, 256, 0, stream>>>(E, w1, H);
    pair_head<<<8, 512, 0, stream>>>(H, b1, w2, b2, out);
}
